// LSTMAutoRegressive_88321707475365
// MI455X (gfx1250) — matrix operand order repaired and checked by reading, not compiled, not run on hardware
//
#include <hip/hip_runtime.h>
#include <hip/hip_bf16.h>

// MI455X (gfx1250, wave32) autoregressive 2-layer LSTM.
// B=1024 sequences, T=1024 serial steps. 64 workgroups x 4 waves; each
// workgroup owns a 16-row batch tile and iterates the 1024-step recurrence
// with v_wmma_f32_16x16x32_f16 for the recurrent GEMMs, LDS round-trips for
// the D-layout -> A-layout transform, register-resident W_hh1 fragments, and
// all activations routed through gfx1250's native v_tanh_f32 TRANS op.

typedef __attribute__((ext_vector_type(16))) _Float16 v16h;
typedef __attribute__((ext_vector_type(8)))  float    v8f;

#define T_STEPS 1024

union FragU {
    v16h     h;
    _Float16 e[16];
    int4     i2[2];
};

// A-fragment (16x32 f16, row-major source in LDS, stride in halves).
// Lane holds M=lane%16, K = kb + hi*16 + {0..15} (matches B-fragment order).
static __device__ __forceinline__ v16h ldsA(const _Float16* M, int stride,
                                            int kbase, int l16, int hi) {
    FragU u;
    const int base = l16 * stride + kbase + hi * 16;
    u.i2[0] = *reinterpret_cast<const int4*>(M + base);
    u.i2[1] = *reinterpret_cast<const int4*>(M + base + 8);
    return u.h;
}

// B-fragment (32x16 f16). Lane holds column n, K = kbase + hi*16 + {0..15}.
static __device__ __forceinline__ v16h ldsB(const _Float16* M, int stride,
                                            int nrow, int kbase, int hi) {
    FragU u;
    const int base = nrow * stride + kbase + hi * 16;
    u.i2[0] = *reinterpret_cast<const int4*>(M + base);
    u.i2[1] = *reinterpret_cast<const int4*>(M + base + 8);
    return u.h;
}

static __device__ __forceinline__ v8f wmma16(v16h a, v16h b, v8f c) {
    return __builtin_amdgcn_wmma_f32_16x16x32_f16(false, a, false, b,
                                                  (short)0, c, false, false);
}

static __device__ __forceinline__ float ftanh(float x) {
#if __has_builtin(__builtin_amdgcn_tanhf)
    return __builtin_amdgcn_tanhf(x);          // gfx1250 v_tanh_f32 (TRANS)
#else
    float e = __builtin_amdgcn_exp2f(-2.885390082f * x);
    return (1.0f - e) * __builtin_amdgcn_rcpf(1.0f + e);
#endif
}

// sigmoid(x) = 0.5*tanh(x/2) + 0.5  ->  1 TRANS + 2 VALU (vs exp+rcp = 2 TRANS)
static __device__ __forceinline__ float fsig(float x) {
#if __has_builtin(__builtin_amdgcn_tanhf)
    return __builtin_fmaf(0.5f, __builtin_amdgcn_tanhf(0.5f * x), 0.5f);
#else
    return __builtin_amdgcn_rcpf(1.0f + __builtin_amdgcn_exp2f(-1.442695041f * x));
#endif
}

#define A1_S 72   // halves; 144B row stride -> conflict-free b128 bursts
#define A2_S 40   // halves; 80B row stride  -> conflict-free
#define W2_S 104  // halves; 208B row stride -> conflict-free

__global__ __launch_bounds__(128, 1)
void lstm_ar_gfx1250(const float* __restrict__ gin,
                     const float* __restrict__ gdel,
                     const float* __restrict__ Wih1, const float* __restrict__ Whh1,
                     const float* __restrict__ bih1, const float* __restrict__ bhh1,
                     const float* __restrict__ Wih2, const float* __restrict__ Whh2,
                     const float* __restrict__ bih2, const float* __restrict__ bhh2,
                     const float* __restrict__ Wout, const float* __restrict__ boutp,
                     float* __restrict__ gout) {
    __shared__ __align__(16) _Float16 sW2[128 * W2_S];   // [Wih2 | Whh2], K=0..95
    __shared__ __align__(16) _Float16 sA1[2][16 * A1_S]; // h1 rows (f16), dbl-buffered
    __shared__ __align__(16) _Float16 sA2[2][16 * A2_S]; // h2 rows (f16), dbl-buffered
    __shared__ float sX[2][32];                          // x0/x1 per row (f32)
    __shared__ float sYP[32];                            // y partials (wave0|wave1)

    const int tid  = threadIdx.x;
    const int wid  = tid >> 5;
    const int lane = tid & 31;
    const int l16  = lane & 15;
    const int hi   = lane >> 4;
    const int b0   = blockIdx.x * 16;

    // ---- Prologue: fill LDS ------------------------------------------------
    for (int idx = tid; idx < 128 * W2_S; idx += 128) {
        int n = idx / W2_S, k = idx - n * W2_S;
        float v = 0.0f;
        if (k < 64)      v = Wih2[n * 64 + k];
        else if (k < 96) v = Whh2[n * 32 + (k - 64)];
        sW2[idx] = (_Float16)v;
    }
    for (int idx = tid; idx < 2 * 16 * A1_S; idx += 128) (&sA1[0][0])[idx] = (_Float16)0.0f;
    for (int idx = tid; idx < 2 * 16 * A2_S; idx += 128) (&sA2[0][0])[idx] = (_Float16)0.0f;
    if (tid < 32) { sX[0][tid] = 0.0f; sX[1][tid] = 0.0f; sYP[tid] = 0.0f; }

    // ---- Persistent per-lane state -----------------------------------------
    // W_hh1 B-fragments: 4 gate tiles x 2 K-steps, register-resident.
    FragU B1[4][2];
#pragma unroll
    for (int gs = 0; gs < 4; ++gs) {
        const int n = (gs * 4 + wid) * 16 + l16;
#pragma unroll
        for (int ks = 0; ks < 2; ++ks) {
            const int kb = ks * 32 + hi * 16;
#pragma unroll
            for (int i = 0; i < 16; ++i)
                B1[gs][ks].e[i] = (_Float16)Whh1[n * 64 + kb + i];
        }
    }
    const int j1 = wid * 16 + l16;
    float bs1[4], wx0[4], wx1[4];
#pragma unroll
    for (int gs = 0; gs < 4; ++gs) {
        const int n = gs * 64 + j1;            // gate index (i,f,g,o blocks)
        bs1[gs] = bih1[n] + bhh1[n];
        const int nr = (gs * 4 + wid) * 16 + l16;
        wx0[gs] = Wih1[nr * 2 + 0];
        wx1[gs] = Wih1[nr * 2 + 1];
    }
    float bs2[4] = {0.f, 0.f, 0.f, 0.f};
    float wv = 0.0f;
    if (wid < 2) {
        const int j2 = wid * 16 + l16;
#pragma unroll
        for (int gs = 0; gs < 4; ++gs) {
            const int n = gs * 32 + j2;
            bs2[gs] = bih2[n] + bhh2[n];
        }
        wv = Wout[wid * 16 + l16];
    }
    const float bout = boutp[0];

    // x[0] into sX[0]
    if (wid == 3) {
        const float xv0 = (lane < 16) ? gin[(size_t)(b0 + l16) * T_STEPS]
                                      : gdel[(size_t)(b0 + l16) * T_STEPS];
        sX[0][l16 * 2 + hi] = xv0;
    }

    float c1r[8], c2r[8];
#pragma unroll
    for (int v = 0; v < 8; ++v) { c1r[v] = 0.0f; c2r[v] = 0.0f; }

    __syncthreads();

    // ---- Recurrence --------------------------------------------------------
    for (int t = 0; t < T_STEPS; ++t) {
        const _Float16* A1r = sA1[t & 1];
        _Float16*       A1w = sA1[(t + 1) & 1];
        const _Float16* A2r = sA2[t & 1];
        _Float16*       A2w = sA2[(t + 1) & 1];
        const float*    sXr = sX[t & 1];

        // wave3: prefetch next-step inputs (global, strided but L2-friendly)
        float xv = 0.0f;
        if (wid == 3 && (t + 1) < T_STEPS) {
            xv = (lane < 16) ? gin[(size_t)(b0 + l16) * T_STEPS + t + 1]
                             : gdel[(size_t)(b0 + l16) * T_STEPS + t + 1];
        }

        // x broadcast for C-init (row m = j + 8*hi)
        float2 xp[8];
#pragma unroll
        for (int j = 0; j < 8; ++j)
            xp[j] = *reinterpret_cast<const float2*>(&sXr[(j + 8 * hi) * 2]);

        // ---- Layer 1: gates = h1·Whh1^T (WMMA) + x·Wih1^T + b (C-init) ----
        v16h a0 = ldsA(A1r, A1_S, 0, l16, hi);
        v16h a1 = ldsA(A1r, A1_S, 32, l16, hi);
        v8f g[4];
#pragma unroll
        for (int gs = 0; gs < 4; ++gs) {
            v8f c;
#pragma unroll
            for (int j = 0; j < 8; ++j)
                c[j] = bs1[gs] + xp[j].x * wx0[gs] + xp[j].y * wx1[gs];
            c = wmma16(a0, B1[gs][0].h, c);
            c = wmma16(a1, B1[gs][1].h, c);
            g[gs] = c;
        }
#pragma unroll
        for (int v = 0; v < 8; ++v) {
            const float cn = fsig(g[1][v]) * c1r[v] + fsig(g[0][v]) * ftanh(g[2][v]);
            const float hn = fsig(g[3][v]) * ftanh(cn);
            c1r[v] = cn;
            A1w[(v + 8 * hi) * A1_S + wid * 16 + l16] = (_Float16)hn;
        }
        if (wid == 3 && (t + 1) < T_STEPS) sX[(t + 1) & 1][l16 * 2 + hi] = xv;

        __syncthreads();   // h1[t] + x[t+1] visible

        // ---- Layer 2 (waves 0,1): gates2 = [h1|h2]·[Wih2|Whh2]^T + b ------
        if (wid < 2) {
            v16h d0 = ldsA(A1w, A1_S, 0, l16, hi);
            v16h d1 = ldsA(A1w, A1_S, 32, l16, hi);
            v16h d2 = ldsA(A2r, A2_S, 0, l16, hi);
            v8f q[4];
#pragma unroll
            for (int gs = 0; gs < 4; ++gs) {
                const int nrow = (gs * 2 + wid) * 16 + l16;
                v8f c;
#pragma unroll
                for (int j = 0; j < 8; ++j) c[j] = bs2[gs];
                c = wmma16(d0, ldsB(sW2, W2_S, nrow, 0, hi), c);
                c = wmma16(d1, ldsB(sW2, W2_S, nrow, 32, hi), c);
                c = wmma16(d2, ldsB(sW2, W2_S, nrow, 64, hi), c);
                q[gs] = c;
            }
            float p[8];
#pragma unroll
            for (int v = 0; v < 8; ++v) {
                const float cn = fsig(q[1][v]) * c2r[v] + fsig(q[0][v]) * ftanh(q[2][v]);
                const float hn = fsig(q[3][v]) * ftanh(cn);
                c2r[v] = cn;
                A2w[(v + 8 * hi) * A2_S + wid * 16 + l16] = (_Float16)hn;
                p[v] = hn * wv;
            }
            // y partial: reduce over the 16 feature lanes of this wave's tile
#pragma unroll
            for (int v = 0; v < 8; ++v) {
                float s = p[v];
                s += __shfl_xor(s, 1);
                s += __shfl_xor(s, 2);
                s += __shfl_xor(s, 4);
                s += __shfl_xor(s, 8);
                p[v] = s;
            }
            if (l16 == 0) {
#pragma unroll
                for (int v = 0; v < 8; ++v) sYP[wid * 16 + v + 8 * hi] = p[v];
            }
        }

        __syncthreads();   // h2[t] + y partials visible

        if (wid == 2 && lane < 16) {
            gout[(size_t)(b0 + lane) * T_STEPS + t] =
                sYP[lane] + sYP[16 + lane] + bout;
        }
    }
}

extern "C" void kernel_launch(void* const* d_in, const int* in_sizes, int n_in,
                              void* d_out, int out_size, void* d_ws, size_t ws_size,
                              hipStream_t stream) {
    const float* input   = (const float*)d_in[0];
    const float* delayed = (const float*)d_in[1];
    const float* Wih1    = (const float*)d_in[2];
    const float* Whh1    = (const float*)d_in[3];
    const float* bih1    = (const float*)d_in[4];
    const float* bhh1    = (const float*)d_in[5];
    const float* Wih2    = (const float*)d_in[6];
    const float* Whh2    = (const float*)d_in[7];
    const float* bih2    = (const float*)d_in[8];
    const float* bhh2    = (const float*)d_in[9];
    const float* Wout    = (const float*)d_in[10];
    const float* bout    = (const float*)d_in[11];
    float* out = (float*)d_out;

    const int B = in_sizes[0] / T_STEPS;   // N_IN == 1
    dim3 grid(B / 16), block(128);
    hipLaunchKernelGGL(lstm_ar_gfx1250, grid, block, 0, stream,
                       input, delayed, Wih1, Whh1, bih1, bhh1,
                       Wih2, Whh2, bih2, bhh2, Wout, bout, out);
}
